// PointerAttention_25288767439145
// MI455X (gfx1250) — matrix operand order repaired and checked by reading, not compiled, not run on hardware
//
#include <hip/hip_runtime.h>
#include <hip/hip_bf16.h>

typedef __bf16 bf16;
typedef __attribute__((ext_vector_type(16))) __bf16 v16bf;
typedef __attribute__((ext_vector_type(8)))  __bf16 v8bf;
typedef __attribute__((ext_vector_type(8)))  float  v8f;

#define DM   512   // d_model == K (contraction depth)
#define TT   2048  // T
#define SS   4096  // S
#define NB   4     // batch
#define BT   128   // block tile (M and N)
#define KT   32    // K tile per WMMA step
#define NKT  (DM / KT)
#define LDST 40    // padded LDS row stride in bf16 elements (32 + 8)

static __device__ __forceinline__ unsigned pack_bf16x2(float x, float y) {
  unsigned short a = __builtin_bit_cast(unsigned short, (__bf16)x);
  unsigned short b = __builtin_bit_cast(unsigned short, (__bf16)y);
  return (unsigned)a | ((unsigned)b << 16);
}

// A-matrix fragment (16x32, M x K): lane holds K {0..7,16..23} or {8..15,24..31}
static __device__ __forceinline__ v16bf frag_a(const bf16* p /* row base + lka */) {
  v8bf lo = *(const v8bf*)p;
  v8bf hi = *(const v8bf*)(p + 16);
  v16bf r;
#pragma unroll
  for (int i = 0; i < 8; ++i) { r[i] = lo[i]; r[i + 8] = hi[i]; }
  return r;
}

// B-matrix fragment (32x16, K x N, stored N-major): lane holds K {0..7,16..23} or {8..15,24..31}
static __device__ __forceinline__ v16bf frag_b(const bf16* p /* row base + half*8 */) {
  v8bf lo = *(const v8bf*)p;
  v8bf hi = *(const v8bf*)(p + 16);
  v16bf r;
#pragma unroll
  for (int i = 0; i < 8; ++i) { r[i] = lo[i]; r[i + 8] = hi[i]; }
  return r;
}

// gfx1250 async copy: 16 bytes global -> LDS per lane, tracked by ASYNCcnt.
// GVS mode: vdst = LDS byte offset (VGPR), vaddr = 32-bit byte offset (VGPR),
// saddr = 64-bit base (SGPR pair).
static __device__ __forceinline__ void async_cp16(unsigned lds_byte,
                                                  unsigned long long gbase,
                                                  unsigned voff) {
  asm volatile("global_load_async_to_lds_b128 %0, %1, %2"
               :: "v"(lds_byte), "v"(voff), "s"(gbase)
               : "memory");
}

static __device__ __forceinline__ void wait_async0() {
  asm volatile("s_wait_asynccnt 0x0" ::: "memory");
}

// ---------------------------------------------------------------------------
// Projection GEMM: out_bf16[m,n] = bf16( sum_k A[m,k]*W[n,k] + bias[n] )
// A fp32 [M,512] (M divisible by 128), W fp32 [512,512] (N x K), bias fp32[512]
// grid: (512/128, M/128), block 256
// ---------------------------------------------------------------------------
__global__ __launch_bounds__(256)
void proj_gemm_bf16(const float* __restrict__ A,
                    const float* __restrict__ W,
                    const float* __restrict__ bias,
                    bf16* __restrict__ out) {
  __shared__ __align__(16) bf16 lA[BT * LDST];
  __shared__ __align__(16) bf16 lB[BT * LDST];

  const int tid   = threadIdx.x;
  const int lane  = tid & 31;
  const int wave  = tid >> 5;
  const int wr    = wave >> 2;        // 0..1 (wave row)
  const int wc    = wave & 3;         // 0..3 (wave col)
  const int lm    = lane & 15;
  const int lhalf = lane >> 4;        // 0/1
  const int lka   = lhalf << 3;       // A frag K base: 0 or 8

  const size_t row0 = (size_t)blockIdx.y * BT;
  const int    col0 = blockIdx.x * BT;

  v8f zero = {};
  v8f acc[4][2];
#pragma unroll
  for (int mi = 0; mi < 4; ++mi)
#pragma unroll
    for (int ni = 0; ni < 2; ++ni) acc[mi][ni] = zero;

  for (int kt = 0; kt < DM; kt += KT) {
    // cooperative load 128x32 fp32 of A and W, convert -> bf16 in LDS
#pragma unroll
    for (int i = 0; i < 4; ++i) {
      int cc = tid + i * 256;         // 0..1023 float4 chunks
      int r  = cc >> 3, c4 = cc & 7;
      float4 va = *(const float4*)(A + (row0 + (size_t)r) * DM + kt + c4 * 4);
      uint2 pa; pa.x = pack_bf16x2(va.x, va.y); pa.y = pack_bf16x2(va.z, va.w);
      *(uint2*)&lA[r * LDST + c4 * 4] = pa;
      float4 vw = *(const float4*)(W + (size_t)(col0 + r) * DM + kt + c4 * 4);
      uint2 pw; pw.x = pack_bf16x2(vw.x, vw.y); pw.y = pack_bf16x2(vw.z, vw.w);
      *(uint2*)&lB[r * LDST + c4 * 4] = pw;
    }
    __syncthreads();

    v16bf af[4];
#pragma unroll
    for (int mi = 0; mi < 4; ++mi)
      af[mi] = frag_a(&lA[(wr * 64 + mi * 16 + lm) * LDST + lka]);
    v16bf bf_[2];
#pragma unroll
    for (int ni = 0; ni < 2; ++ni)
      bf_[ni] = frag_b(&lB[(wc * 32 + ni * 16 + lm) * LDST + lhalf * 8]);

#pragma unroll
    for (int mi = 0; mi < 4; ++mi)
#pragma unroll
      for (int ni = 0; ni < 2; ++ni)
        acc[mi][ni] = __builtin_amdgcn_wmma_f32_16x16x32_bf16(
            false, af[mi], false, bf_[ni], (short)0, acc[mi][ni], false, false);
    __syncthreads();
  }

  // epilogue: +bias, convert to bf16, store
#pragma unroll
  for (int ni = 0; ni < 2; ++ni) {
    const int n = col0 + wc * 32 + ni * 16 + lm;
    const float bv = bias[n];
#pragma unroll
    for (int mi = 0; mi < 4; ++mi) {
#pragma unroll
      for (int r = 0; r < 8; ++r) {
        size_t m = row0 + wr * 64 + mi * 16 + r + 8 * lhalf;
        out[m * DM + n] = (bf16)(acc[mi][ni][r] + bv);
      }
    }
  }
}

// ---------------------------------------------------------------------------
// Main QK^T GEMM: out[b,t,s] = mask[b,s] ? -inf : (1/64) * sum_d qb[b,t,d]*kb[b,s,d]
// Double-buffered LDS with gfx1250 async global->LDS copies.
// grid: (S/128, T/128, B), block 256
// ---------------------------------------------------------------------------
__global__ __launch_bounds__(256)
void qk_logits(const bf16* __restrict__ qb,            // [B*T, 512]
               const bf16* __restrict__ kb,            // [B*S, 512]
               const unsigned char* __restrict__ mask, // [B, S] bool
               float* __restrict__ out) {              // [B, T, S]
  __shared__ __align__(16) bf16 ldsA[2][BT * LDST];
  __shared__ __align__(16) bf16 ldsB[2][BT * LDST];

  const int tid   = threadIdx.x;
  const int lane  = tid & 31;
  const int wave  = tid >> 5;
  const int wr    = wave >> 2;
  const int wc    = wave & 3;
  const int lm    = lane & 15;
  const int lhalf = lane >> 4;
  const int lka   = lhalf << 3;

  const int b    = blockIdx.z;
  const int row0 = blockIdx.y * BT;   // T tile
  const int col0 = blockIdx.x * BT;   // S tile

  const unsigned long long Aq = (unsigned long long)(qb + ((size_t)b * TT + row0) * DM);
  const unsigned long long Bk = (unsigned long long)(kb + ((size_t)b * SS + col0) * DM);

  // LDS byte offsets of the two buffers (low 32 bits of generic address)
  unsigned aA[2], aB[2];
  aA[0] = (unsigned)(size_t)&ldsA[0][0];
  aA[1] = (unsigned)(size_t)&ldsA[1][0];
  aB[0] = (unsigned)(size_t)&ldsB[0][0];
  aB[1] = (unsigned)(size_t)&ldsB[1][0];

  // per-thread chunk coords: 512 16B chunks per 128x32 tile, 2 per thread
  const int r0 = tid >> 2, c0 = (tid & 3) * 16;          // chunk 0
  const int r1 = (tid + 256) >> 2, c1 = ((tid + 256) & 3) * 16; // chunk 1
  const unsigned l0 = (unsigned)(r0 * (LDST * 2)) + c0;  // LDS byte offsets
  const unsigned l1 = (unsigned)(r1 * (LDST * 2)) + c1;
  const unsigned g0 = (unsigned)(r0 * (DM * 2)) + c0;    // global byte offsets (kt=0)
  const unsigned g1 = (unsigned)(r1 * (DM * 2)) + c1;

  v8f zero = {};
  v8f acc[4][2];
#pragma unroll
  for (int mi = 0; mi < 4; ++mi)
#pragma unroll
    for (int ni = 0; ni < 2; ++ni) acc[mi][ni] = zero;

  // prologue: async-load k-tile 0 into buffer 0
  async_cp16(aA[0] + l0, Aq, g0);
  async_cp16(aA[0] + l1, Aq, g1);
  async_cp16(aB[0] + l0, Bk, g0);
  async_cp16(aB[0] + l1, Bk, g1);

  for (int it = 0; it < NKT; ++it) {
    wait_async0();      // this wave's copies for tile `it` are done
    __syncthreads();    // => whole tile `it` visible; all waves done reading buf (it+1)&1

    const int cb = it & 1;
    if (it + 1 < NKT) {
      const int nb = (it + 1) & 1;
      const unsigned ko = (unsigned)((it + 1) * KT * 2); // byte advance in K
      async_cp16(aA[nb] + l0, Aq, g0 + ko);
      async_cp16(aA[nb] + l1, Aq, g1 + ko);
      async_cp16(aB[nb] + l0, Bk, g0 + ko);
      async_cp16(aB[nb] + l1, Bk, g1 + ko);
    }

    const bf16* tA = ldsA[cb];
    const bf16* tB = ldsB[cb];
    v16bf af[4];
#pragma unroll
    for (int mi = 0; mi < 4; ++mi)
      af[mi] = frag_a(&tA[(wr * 64 + mi * 16 + lm) * LDST + lka]);
    v16bf bf_[2];
#pragma unroll
    for (int ni = 0; ni < 2; ++ni)
      bf_[ni] = frag_b(&tB[(wc * 32 + ni * 16 + lm) * LDST + lhalf * 8]);

#pragma unroll
    for (int mi = 0; mi < 4; ++mi)
#pragma unroll
      for (int ni = 0; ni < 2; ++ni)
        acc[mi][ni] = __builtin_amdgcn_wmma_f32_16x16x32_bf16(
            false, af[mi], false, bf_[ni], (short)0, acc[mi][ni], false, false);
  }

  const float scale = 1.0f / 64.0f;   // 1 / (N_HEADS * sqrt(HEAD_DIM))
  float* outb = out + (size_t)b * TT * SS;
#pragma unroll
  for (int ni = 0; ni < 2; ++ni) {
    const int n = col0 + wc * 32 + ni * 16 + lm;
    const bool mk = mask[(size_t)b * SS + n] != 0;
#pragma unroll
    for (int mi = 0; mi < 4; ++mi) {
#pragma unroll
      for (int r = 0; r < 8; ++r) {
        int m = row0 + wr * 64 + mi * 16 + r + 8 * lhalf;
        float v = acc[mi][ni][r] * scale;
        // write-once 128MB output: non-temporal store, keep qb/kb resident in L2
        __builtin_nontemporal_store(mk ? -__builtin_inff() : v,
                                    &outb[(size_t)m * SS + n]);
      }
    }
  }
}

// ---------------------------------------------------------------------------
extern "C" void kernel_launch(void* const* d_in, const int* in_sizes, int n_in,
                              void* d_out, int out_size, void* d_ws, size_t ws_size,
                              hipStream_t stream) {
  const float* query = (const float*)d_in[0];              // [4,2048,512]
  const float* keys  = (const float*)d_in[1];              // [4,4096,512]
  const unsigned char* mask = (const unsigned char*)d_in[2]; // [4,4096] bool
  const float* q_w = (const float*)d_in[3];                // [512,512]
  const float* q_b = (const float*)d_in[4];                // [512]
  const float* k_w = (const float*)d_in[5];                // [512,512]
  const float* k_b = (const float*)d_in[6];                // [512]
  float* out = (float*)d_out;

  // workspace: bf16 projected activations (8 MB + 16 MB)
  bf16* qbuf = (bf16*)d_ws;
  bf16* kbuf = (bf16*)((char*)d_ws + (size_t)NB * TT * DM * sizeof(bf16));

  dim3 blk(256);
  proj_gemm_bf16<<<dim3(DM / BT, (NB * TT) / BT, 1), blk, 0, stream>>>(query, q_w, q_b, qbuf);
  proj_gemm_bf16<<<dim3(DM / BT, (NB * SS) / BT, 1), blk, 0, stream>>>(keys, k_w, k_b, kbuf);
  qk_logits<<<dim3(SS / BT, TT / BT, NB), blk, 0, stream>>>(qbuf, kbuf, mask, out);
}
